// GraphGDP_13022340841832
// MI455X (gfx1250) — matrix operand order repaired and checked by reading, not compiled, not run on hardware
//
#include <hip/hip_runtime.h>

typedef __attribute__((ext_vector_type(16))) _Float16 v16h;
typedef __attribute__((ext_vector_type(8)))  _Float16 v8h;
typedef __attribute__((ext_vector_type(8)))  float    v8f;
typedef __attribute__((ext_vector_type(4)))  float    v4f;

#define HD 256

// ---------------------------------------------------------------------------
// Generic WMMA GEMM: C[R x 256] = act(A[R x K] @ W + bias)
// W is pre-transposed to f16: Wt[n][k] (n = output col, row-major, ld=K).
// GATHER=true composes A rows on the fly as [x[src]|x[dst]|e1] (K=1280).
// One wave -> 16x64 tile; 128-thread block covers 16x256.
// ---------------------------------------------------------------------------
template <bool GATHER>
__global__ void __launch_bounds__(128)
k_gemm256(const float* __restrict__ A, int lda, int K,
          const _Float16* __restrict__ Wt,
          const float* __restrict__ bias,
          float* __restrict__ C, int ldc, int act,
          const int* __restrict__ gsrc, const int* __restrict__ gdst,
          const float* __restrict__ gx, const float* __restrict__ ge)
{
    const int lane = threadIdx.x & 31;
    const int wave = threadIdx.x >> 5;
    const int mr   = blockIdx.x * 16;
    const int rlo  = lane & 15;
    const bool hi  = lane >= 16;
    const int rowA = mr + rlo;

    int s0 = 0, s1 = 0;
    if (GATHER) { s0 = gsrc[rowA]; s1 = gdst[rowA]; }

    v8f acc[4] = {};
    for (int kc = 0; kc < K; kc += 32) {
        // ---- A fragment (16x32 f16): lanes 0-15 K={kc..+7,kc+16..+23},
        //      lanes 16-31 shifted by +8.
        const int kb = kc + (hi ? 8 : 0);
        const float* ap;
        if (GATHER) {
            if (kc < 512)       ap = gx + (size_t)s0 * 512 + kb;
            else if (kc < 1024) ap = gx + (size_t)s1 * 512 + (kb - 512);
            else                ap = ge + (size_t)rowA * HD + (kb - 1024);
        } else {
            ap = A + (size_t)rowA * lda + kb;
        }
        v4f a0 = *(const v4f*)(ap);
        v4f a1 = *(const v4f*)(ap + 4);
        v4f a2 = *(const v4f*)(ap + 16);
        v4f a3 = *(const v4f*)(ap + 20);
        v16h af;
#pragma unroll
        for (int i = 0; i < 4; ++i) {
            af[i]      = (_Float16)a0[i];
            af[4 + i]  = (_Float16)a1[i];
            af[8 + i]  = (_Float16)a2[i];
            af[12 + i] = (_Float16)a3[i];
        }
        // ---- B fragments: same K order as A: lanes 0-15 K={kc..+7,kc+16..+23},
        //      lanes 16-31 shifted by +8, from transposed weights.
        const int kbB = kc + (hi ? 8 : 0);
#pragma unroll
        for (int t = 0; t < 4; ++t) {
            const int n = (wave << 6) + (t << 4) + rlo;
            const _Float16* wp = Wt + (size_t)n * K + kbB;
            v8h b0 = *(const v8h*)wp;
            v8h b1 = *(const v8h*)(wp + 16);
            v16h bf;
#pragma unroll
            for (int i = 0; i < 8; ++i) { bf[i] = b0[i]; bf[8 + i] = b1[i]; }
            acc[t] = __builtin_amdgcn_wmma_f32_16x16x32_f16(
                false, af, false, bf, (short)0, acc[t], false, false);
        }
    }
    // ---- epilogue: lanes 0-15 hold M=r, lanes 16-31 hold M=8+r
#pragma unroll
    for (int t = 0; t < 4; ++t) {
        const int n = (wave << 6) + (t << 4) + rlo;
        const float bv = bias ? bias[n] : 0.0f;
#pragma unroll
        for (int r = 0; r < 8; ++r) {
            const int row = mr + r + (hi ? 8 : 0);
            float v = acc[t][r] + bv;
            if (act == 1)      v = v > 0.0f ? v : 0.0f;
            else if (act == 2) v = v > 0.0f ? v : 0.2f * v;
            C[(size_t)row * ldc + n] = v;
        }
    }
}

// ---------------------------------------------------------------------------
// Weight fp32 (K x M) -> f16 transposed (M x K)
// ---------------------------------------------------------------------------
__global__ void k_w2h(const float* __restrict__ W, _Float16* __restrict__ Wt,
                      int K, int M)
{
    int idx = blockIdx.x * blockDim.x + threadIdx.x;
    if (idx >= K * M) return;
    int k = idx / M, n = idx % M;
    Wt[(size_t)n * K + k] = (_Float16)W[(size_t)k * M + n];
}

// ---------------------------------------------------------------------------
// Time-encoder MLP for B<=16 scalar t-values (tiny; one block of 256)
// ---------------------------------------------------------------------------
__global__ void __launch_bounds__(256)
k_tenc(const float* __restrict__ t,
       const float* __restrict__ W0, const float* __restrict__ b0,
       const float* __restrict__ W1, const float* __restrict__ b1,
       const float* __restrict__ W2, const float* __restrict__ b2,
       float* __restrict__ tenc, int Bn)
{
    __shared__ float h0[16][HD];
    __shared__ float h1[16][HD];
    const int j = threadIdx.x;
    for (int b = 0; b < Bn; ++b) {
        float v = t[b] * W0[j] + b0[j];
        h0[b][j] = v > 0.0f ? v : 0.0f;
    }
    __syncthreads();
    for (int b = 0; b < Bn; ++b) {
        float s = b1[j];
        for (int k = 0; k < HD; ++k) s += h0[b][k] * W1[k * HD + j];
        h1[b][j] = s > 0.0f ? s : 0.0f;
    }
    __syncthreads();
    for (int b = 0; b < Bn; ++b) {
        float s = b2[j];
        for (int k = 0; k < HD; ++k) s += h1[b][k] * W2[k * HD + j];
        tenc[b * HD + j] = s;
    }
}

// x[n] = concat(tenc[batch[n]], tenc[batch[n]])
__global__ void __launch_bounds__(256)
k_scatter_x(const float* __restrict__ tenc, const int* __restrict__ batch,
            float* __restrict__ x)
{
    const int n = blockIdx.x, j = threadIdx.x;
    const float v = tenc[batch[n] * HD + j];
    x[(size_t)n * 512 + j]       = v;
    x[(size_t)n * 512 + HD + j]  = v;
}

// first edge-encoder layer: h0[e][j] = relu(a[e]*W0[j] + b0[j])
__global__ void k_edge_l0(const float* __restrict__ a,
                          const float* __restrict__ W0,
                          const float* __restrict__ b0,
                          float* __restrict__ h0, int E)
{
    int idx = blockIdx.x * blockDim.x + threadIdx.x;
    if (idx >= E * HD) return;
    int e = idx >> 8, j = idx & 255;
    float v = a[e] * W0[j] + b0[j];
    h0[idx] = v > 0.0f ? v : 0.0f;
}

__global__ void k_fill_f32(float* p, float v, int n)
{
    int i = blockIdx.x * blockDim.x + threadIdx.x;
    if (i < n) p[i] = v;
}
__global__ void k_fill_u32(unsigned* p, unsigned v, int n)
{
    int i = blockIdx.x * blockDim.x + threadIdx.x;
    if (i < n) p[i] = v;
}

// x[n][j]=bias1[j]; x[n][256+j]=bias2[j]  (GAT output bias init)
__global__ void __launch_bounds__(256)
k_init_xbias(float* __restrict__ x, const float* __restrict__ b1,
             const float* __restrict__ b2)
{
    const int n = blockIdx.x, j = threadIdx.x;
    x[(size_t)n * 512 + j]      = b1[j];
    x[(size_t)n * 512 + HD + j] = b2[j];
}

__device__ __forceinline__ unsigned f32_sortable(float f)
{
    unsigned u = __float_as_uint(f);
    return (u & 0x80000000u) ? ~u : (u | 0x80000000u);
}
__device__ __forceinline__ float sortable_f32(unsigned u)
{
    return __uint_as_float((u & 0x80000000u) ? (u ^ 0x80000000u) : ~u);
}

// per-edge attention logits + segment max (wave per edge)
__global__ void __launch_bounds__(256)
k_logits(const float* __restrict__ xl, const float* __restrict__ xr,
         const float* __restrict__ eW,
         const int* __restrict__ src, const int* __restrict__ dst,
         const float* __restrict__ att,
         float* __restrict__ logits, unsigned* __restrict__ segmax, int E)
{
    const int e = blockIdx.x * 8 + (threadIdx.x >> 5);
    if (e >= E) return;
    const int lane = threadIdx.x & 31;
    const int s = src[e], d = dst[e];
    float acc = 0.0f;
#pragma unroll
    for (int i = 0; i < 8; ++i) {
        const int j = lane + i * 32;
        float m = xl[(size_t)s * HD + j] + xr[(size_t)d * HD + j] +
                  eW[(size_t)e * HD + j];
        m = m > 0.0f ? m : 0.2f * m;          // LeakyReLU(0.2)
        acc += att[j] * m;
    }
    for (int o = 16; o; o >>= 1) acc += __shfl_xor(acc, o);
    if (lane == 0) {
        logits[e] = acc;
        atomicMax(segmax + d, f32_sortable(acc));
    }
}

__global__ void k_exp(const float* __restrict__ logits,
                      const int* __restrict__ dst,
                      const unsigned* __restrict__ segmax,
                      float* __restrict__ ex, float* __restrict__ denom, int E)
{
    int e = blockIdx.x * blockDim.x + threadIdx.x;
    if (e >= E) return;
    const int d = dst[e];
    const float m = sortable_f32(segmax[d]);
    const float v = __expf(logits[e] - m);
    ex[e] = v;
    atomicAdd(denom + d, v);
}

// scatter-add: x[dst][coloff+j] += xl[src][j] * alpha  (wave per edge)
__global__ void __launch_bounds__(256)
k_scatter_att(const float* __restrict__ xl,
              const int* __restrict__ src, const int* __restrict__ dst,
              const float* __restrict__ ex, const float* __restrict__ denom,
              float* __restrict__ xout, int coloff, int E)
{
    const int e = blockIdx.x * 8 + (threadIdx.x >> 5);
    if (e >= E) return;
    const int lane = threadIdx.x & 31;
    const int s = src[e], d = dst[e];
    const float alpha = ex[e] / (denom[d] + 1e-16f);
#pragma unroll
    for (int i = 0; i < 8; ++i) {
        const int j = lane + i * 32;
        atomicAdd(&xout[(size_t)d * 512 + coloff + j],
                  xl[(size_t)s * HD + j] * alpha);
    }
}

// final decoder layer (256 -> 1): wave per row
__global__ void __launch_bounds__(256)
k_dec2(const float* __restrict__ h, const float* __restrict__ W2,
       const float* __restrict__ b2, float* __restrict__ out, int E)
{
    const int r = blockIdx.x * 8 + (threadIdx.x >> 5);
    if (r >= E) return;
    const int lane = threadIdx.x & 31;
    float s = 0.0f;
#pragma unroll
    for (int i = 0; i < 8; ++i) {
        const int j = lane + i * 32;
        s += h[(size_t)r * HD + j] * W2[j];
    }
    for (int o = 16; o; o >>= 1) s += __shfl_xor(s, o);
    if (lane == 0) out[r] = s + b2[0];
}

// ---------------------------------------------------------------------------
extern "C" void kernel_launch(void* const* d_in, const int* in_sizes, int n_in,
                              void* d_out, int out_size, void* d_ws, size_t ws_size,
                              hipStream_t stream)
{
    if (n_in < 38) return;
    const int* eidx1   = (const int*)d_in[0];
    const float* eat1  = (const float*)d_in[1];
    const int* eidx2   = (const int*)d_in[2];
    const float* eat2  = (const float*)d_in[3];
    const int* batch   = (const int*)d_in[4];
    const float* tval  = (const float*)d_in[5];
    const float* te_W0 = (const float*)d_in[6],  *te_b0 = (const float*)d_in[7];
    const float* te_W1 = (const float*)d_in[8],  *te_b1 = (const float*)d_in[9];
    const float* te_W2 = (const float*)d_in[10], *te_b2 = (const float*)d_in[11];
    const float* ee_W0 = (const float*)d_in[12], *ee_b0 = (const float*)d_in[13];
    const float* ee_W1 = (const float*)d_in[14], *ee_b1 = (const float*)d_in[15];
    const float* ee_W2 = (const float*)d_in[16], *ee_b2 = (const float*)d_in[17];
    const float* dec_W0 = (const float*)d_in[18], *dec_b0 = (const float*)d_in[19];
    const float* dec_W1 = (const float*)d_in[20], *dec_b1 = (const float*)d_in[21];
    const float* dec_W2 = (const float*)d_in[22], *dec_b2 = (const float*)d_in[23];
    const float* gg_Wl = (const float*)d_in[24], *gg_bl = (const float*)d_in[25];
    const float* gg_Wr = (const float*)d_in[26], *gg_br = (const float*)d_in[27];
    const float* gg_We = (const float*)d_in[28], *gg_att = (const float*)d_in[29];
    const float* gg_bias = (const float*)d_in[30];
    const float* gf_Wl = (const float*)d_in[31], *gf_bl = (const float*)d_in[32];
    const float* gf_Wr = (const float*)d_in[33], *gf_br = (const float*)d_in[34];
    const float* gf_We = (const float*)d_in[35], *gf_att = (const float*)d_in[36];
    const float* gf_bias = (const float*)d_in[37];

    const int E1 = in_sizes[1];
    const int E2 = in_sizes[3];
    const int N  = in_sizes[4];
    const int Bn = in_sizes[5];
    const int L  = 3;

    const int* src1 = eidx1, *dst1 = eidx1 + E1;
    const int* src2 = eidx2, *dst2 = eidx2 + E2;

    // ---- workspace bump allocator ----
    size_t off = 0;
    auto alloc = [&](size_t bytes) -> void* {
        void* p = (void*)((char*)d_ws + off);
        off = (off + bytes + 255) & ~(size_t)255;
        return p;
    };
    _Float16* eeW1T = (_Float16*)alloc((size_t)HD * HD * 2);
    _Float16* eeW2T = (_Float16*)alloc((size_t)HD * HD * 2);
    _Float16* ggWlT = (_Float16*)alloc((size_t)L * 512 * HD * 2);
    _Float16* ggWrT = (_Float16*)alloc((size_t)L * 512 * HD * 2);
    _Float16* gfWlT = (_Float16*)alloc((size_t)L * 512 * HD * 2);
    _Float16* gfWrT = (_Float16*)alloc((size_t)L * 512 * HD * 2);
    _Float16* ggWeT = (_Float16*)alloc((size_t)L * HD * HD * 2);
    _Float16* gfWeT = (_Float16*)alloc((size_t)L * HD * HD * 2);
    _Float16* dW0T  = (_Float16*)alloc((size_t)1280 * HD * 2);
    _Float16* dW1T  = (_Float16*)alloc((size_t)HD * HD * 2);

    float* tenc = (float*)alloc((size_t)16 * HD * 4);
    float* x    = (float*)alloc((size_t)N * 512 * 4);
    float* xl1  = (float*)alloc((size_t)N * HD * 4);
    float* xr1  = (float*)alloc((size_t)N * HD * 4);
    float* xl2  = (float*)alloc((size_t)N * HD * 4);
    float* xr2  = (float*)alloc((size_t)N * HD * 4);
    float* e1   = (float*)alloc((size_t)E1 * HD * 4);
    float* e2   = (float*)alloc((size_t)E2 * HD * 4);
    float* eW1  = (float*)alloc((size_t)E1 * HD * 4);
    float* eW2  = (float*)alloc((size_t)E2 * HD * 4);
    float* tmp  = (float*)alloc((size_t)E1 * HD * 4);
    float* logits = (float*)alloc((size_t)E1 * 4);
    float* ex     = (float*)alloc((size_t)E1 * 4);
    unsigned* segmax = (unsigned*)alloc((size_t)N * 4);
    float* denom     = (float*)alloc((size_t)N * 4);

    // ---- 1) weight transpose/convert to f16 ----
    auto w2h = [&](const float* W, _Float16* Wt, int K, int M) {
        int n = K * M;
        k_w2h<<<(n + 255) / 256, 256, 0, stream>>>(W, Wt, K, M);
    };
    w2h(ee_W1, eeW1T, HD, HD);
    w2h(ee_W2, eeW2T, HD, HD);
    for (int l = 0; l < L; ++l) {
        w2h(gg_Wl + (size_t)l * 512 * HD, ggWlT + (size_t)l * HD * 512, 512, HD);
        w2h(gg_Wr + (size_t)l * 512 * HD, ggWrT + (size_t)l * HD * 512, 512, HD);
        w2h(gf_Wl + (size_t)l * 512 * HD, gfWlT + (size_t)l * HD * 512, 512, HD);
        w2h(gf_Wr + (size_t)l * 512 * HD, gfWrT + (size_t)l * HD * 512, 512, HD);
        w2h(gg_We + (size_t)l * HD * HD, ggWeT + (size_t)l * HD * HD, HD, HD);
        w2h(gf_We + (size_t)l * HD * HD, gfWeT + (size_t)l * HD * HD, HD, HD);
    }
    w2h(dec_W0, dW0T, 1280, HD);
    w2h(dec_W1, dW1T, HD, HD);

    auto gemm = [&](const float* A, int lda, int K, const _Float16* Wt,
                    const float* bias, float* C, int R, int act) {
        k_gemm256<false><<<R / 16, 128, 0, stream>>>(
            A, lda, K, Wt, bias, C, HD, act, nullptr, nullptr, nullptr, nullptr);
    };

    // ---- 2) node features: time MLP -> x = [tenc|tenc] ----
    k_tenc<<<1, 256, 0, stream>>>(tval, te_W0, te_b0, te_W1, te_b1, te_W2, te_b2,
                                  tenc, Bn);
    k_scatter_x<<<N, 256, 0, stream>>>(tenc, batch, x);

    // ---- 3) edge encodings (shared MLP weights) ----
    k_edge_l0<<<(E1 * HD + 255) / 256, 256, 0, stream>>>(eat1, ee_W0, ee_b0, tmp, E1);
    gemm(tmp, HD, HD, eeW1T, ee_b1, eW1, E1, 1);
    gemm(eW1, HD, HD, eeW2T, ee_b2, e1, E1, 0);
    k_edge_l0<<<(E2 * HD + 255) / 256, 256, 0, stream>>>(eat2, ee_W0, ee_b0, tmp, E2);
    gemm(tmp, HD, HD, eeW1T, ee_b1, eW2, E2, 1);
    gemm(eW2, HD, HD, eeW2T, ee_b2, e2, E2, 0);

    // ---- 4) GATv2 layers ----
    for (int l = 0; l < L; ++l) {
        gemm(x, 512, 512, ggWlT + (size_t)l * HD * 512, gg_bl + l * HD, xl1, N, 0);
        gemm(x, 512, 512, ggWrT + (size_t)l * HD * 512, gg_br + l * HD, xr1, N, 0);
        gemm(x, 512, 512, gfWlT + (size_t)l * HD * 512, gf_bl + l * HD, xl2, N, 0);
        gemm(x, 512, 512, gfWrT + (size_t)l * HD * 512, gf_br + l * HD, xr2, N, 0);
        gemm(e1, HD, HD, ggWeT + (size_t)l * HD * HD, nullptr, eW1, E1, 0);
        gemm(e2, HD, HD, gfWeT + (size_t)l * HD * HD, nullptr, eW2, E2, 0);

        k_init_xbias<<<N, 256, 0, stream>>>(x, gg_bias + l * HD, gf_bias + l * HD);

        // graph 1
        k_fill_u32<<<(N + 255) / 256, 256, 0, stream>>>(segmax, 0u, N);
        k_fill_f32<<<(N + 255) / 256, 256, 0, stream>>>(denom, 0.0f, N);
        k_logits<<<(E1 + 7) / 8, 256, 0, stream>>>(xl1, xr1, eW1, src1, dst1,
                                                   gg_att + l * HD, logits, segmax, E1);
        k_exp<<<(E1 + 255) / 256, 256, 0, stream>>>(logits, dst1, segmax, ex, denom, E1);
        k_scatter_att<<<(E1 + 7) / 8, 256, 0, stream>>>(xl1, src1, dst1, ex, denom,
                                                        x, 0, E1);
        // graph 2
        k_fill_u32<<<(N + 255) / 256, 256, 0, stream>>>(segmax, 0u, N);
        k_fill_f32<<<(N + 255) / 256, 256, 0, stream>>>(denom, 0.0f, N);
        k_logits<<<(E2 + 7) / 8, 256, 0, stream>>>(xl2, xr2, eW2, src2, dst2,
                                                   gf_att + l * HD, logits, segmax, E2);
        k_exp<<<(E2 + 255) / 256, 256, 0, stream>>>(logits, dst2, segmax, ex, denom, E2);
        k_scatter_att<<<(E2 + 7) / 8, 256, 0, stream>>>(xl2, src2, dst2, ex, denom,
                                                        x, HD, E2);
    }

    // ---- 5) decoder: fused gather [x[src]|x[dst]|e1] @ W0 -> relu -> W1 -> relu -> W2
    k_gemm256<true><<<E1 / 16, 128, 0, stream>>>(
        nullptr, 0, 1280, dW0T, dec_b0, tmp, HD, 1, src1, dst1, x, e1);
    gemm(tmp, HD, HD, dW1T, dec_b1, eW1, E1, 1);
    k_dec2<<<(E1 + 7) / 8, 256, 0, stream>>>(eW1, dec_W2, dec_b2, (float*)d_out, E1);
}
